// CotSR_18236431139231
// MI455X (gfx1250) — matrix operand order repaired and checked by reading, not compiled, not run on hardware
//
#include <hip/hip_runtime.h>
#include <math.h>

#define BB 4
#define CC 512
#define NN 4096
#define DQV 64

typedef __bf16 bf16_t;
typedef __attribute__((ext_vector_type(16))) __bf16 v16bf;
typedef __attribute__((ext_vector_type(8)))  float  v8f;
typedef __attribute__((ext_vector_type(4)))  float  f32x4;
typedef __attribute__((ext_vector_type(4)))  unsigned int u32x4;
typedef __attribute__((ext_vector_type(8)))  unsigned int u32x8;

__device__ __forceinline__ v8f zero8() {
  v8f z;
#pragma unroll
  for (int i = 0; i < 8; ++i) z[i] = 0.0f;
  return z;
}

// Gather two 16-byte runs into one 32-byte WMMA operand register set.
__device__ __forceinline__ v16bf load_op(const bf16_t* p0, const bf16_t* p1) {
  u32x4 a = *(const u32x4*)p0;
  u32x4 b = *(const u32x4*)p1;
  u32x8 r;
  r[0] = a[0]; r[1] = a[1]; r[2] = a[2]; r[3] = a[3];
  r[4] = b[0]; r[5] = b[1]; r[6] = b[2]; r[7] = b[3];
  return __builtin_bit_cast(v16bf, r);
}

__device__ __forceinline__ v8f wmma_bf16(v16bf a, v16bf b, v8f c) {
  return __builtin_amdgcn_wmma_f32_16x16x32_bf16(false, a, false, b, (short)0, c,
                                                 false, false);
}

// ---------------- weight f32 -> bf16 convert ----------------
__global__ void k_cvt_bf16(const float* __restrict__ in, bf16_t* __restrict__ out, int n) {
  int i = blockIdx.x * blockDim.x + threadIdx.x;
  if (i < n) out[i] = (bf16_t)in[i];
}

// ---------------- x [B][C][N] f32 -> Xt [B][N][C] bf16 ----------------
__global__ void k_transpose_cvt(const float* __restrict__ x, bf16_t* __restrict__ xt) {
  __shared__ bf16_t tile[32][33];
  int b  = blockIdx.z;
  int cb = blockIdx.y * 32;
  int nb = blockIdx.x * 32;
  int tx = threadIdx.x & 31;
  int ty = threadIdx.x >> 5;  // 0..7
#pragma unroll
  for (int i = 0; i < 4; ++i) {
    int c = cb + ty + i * 8;
    tile[ty + i * 8][tx] = (bf16_t)x[((size_t)b * CC + c) * NN + nb + tx];
  }
  __syncthreads();
#pragma unroll
  for (int i = 0; i < 4; ++i) {
    int n = nb + ty + i * 8;
    xt[((size_t)b * NN + n) * CC + cb + tx] = tile[tx][ty + i * 8];
  }
}

// ---- GEMM helpers: operand tile loads + 8-WMMA pack ----
__device__ __forceinline__ void gemm_loads(const bf16_t* const* arow,
                                           const bf16_t* const* brow,
                                           int k, v16bf a[2], v16bf b[4]) {
#pragma unroll
  for (int ms = 0; ms < 2; ++ms) a[ms] = load_op(arow[ms] + k, arow[ms] + k + 16);
#pragma unroll
  for (int cs = 0; cs < 4; ++cs) b[cs] = load_op(brow[cs] + k, brow[cs] + k + 16);
}

__device__ __forceinline__ void gemm_mma(const v16bf a[2], const v16bf b[4], v8f acc[2][4]) {
#pragma unroll
  for (int ms = 0; ms < 2; ++ms)
#pragma unroll
    for (int cs = 0; cs < 4; ++cs)
      acc[ms][cs] = wmma_bf16(a[ms], b[cs], acc[ms][cs]);
}

// ---------------- generic TN GEMM: D[M][Nout] = A[M][K] * Bw[Nout][K]^T + bias ----
// A row-major (K contiguous), Bw row-major (K contiguous) == B column-major.
// Wave tile 32(M) x 64(N). K-loop manually unrolled x2 with ping-pong operand
// buffers: loads for stage i+1 are issued before the WMMA pack of stage i and
// consumed after it, so the compiler can use non-zero loadcnt waits.
// Requires Kdim % 64 == 0.
// bias_axis: 0 -> bias indexed by output column, 1 -> by output row.
__global__ __launch_bounds__(128) void k_gemm_tn(
    const bf16_t* __restrict__ A, size_t strideA,
    const bf16_t* __restrict__ Bw, size_t strideB,
    const float* __restrict__ bias,
    bf16_t* __restrict__ D, size_t strideD,
    int M, int Nout, int Kdim, int bias_axis) {
  int batch = blockIdx.y;
  A  += (size_t)batch * strideA;
  Bw += (size_t)batch * strideB;
  D  += (size_t)batch * strideD;

  int wave = threadIdx.x >> 5;
  int lane = threadIdx.x & 31;
  int strip = blockIdx.x * 4 + wave;
  int stripsPerRow = Nout >> 6;
  int m0 = (strip / stripsPerRow) << 5;  // 32 rows / strip
  int n0 = (strip % stripsPerRow) << 6;  // 64 cols / strip
  if (m0 >= M) return;
  int half = lane >> 4, lm = lane & 15;

  const bf16_t* arow[2];
  arow[0] = A + (size_t)(m0 + lm) * Kdim + half * 8;
  arow[1] = A + (size_t)(m0 + 16 + lm) * Kdim + half * 8;
  const bf16_t* brow[4];
#pragma unroll
  for (int cs = 0; cs < 4; ++cs)
    brow[cs] = Bw + (size_t)(n0 + cs * 16 + lm) * Kdim + half * 8;

  v8f acc[2][4];
#pragma unroll
  for (int ms = 0; ms < 2; ++ms)
#pragma unroll
    for (int cs = 0; cs < 4; ++cs) acc[ms][cs] = zero8();

  v16bf a0[2], b0[4], a1[2], b1[4];
  gemm_loads(arow, brow, 0, a0, b0);
  for (int k0 = 0; k0 < Kdim; k0 += 64) {
    gemm_loads(arow, brow, k0 + 32, a1, b1);       // stage-1 operands in flight
    gemm_mma(a0, b0, acc);                          // consume stage-0
    int k2 = (k0 + 64 < Kdim) ? k0 + 64 : 0;        // clamp: final prefetch discarded
    gemm_loads(arow, brow, k2, a0, b0);             // stage-0' operands in flight
    gemm_mma(a1, b1, acc);                          // consume stage-1
  }

#pragma unroll
  for (int ms = 0; ms < 2; ++ms)
#pragma unroll
    for (int cs = 0; cs < 4; ++cs) {
      int col = n0 + cs * 16 + lm;
      float bc = (bias_axis == 0) ? bias[col] : 0.0f;
#pragma unroll
      for (int r = 0; r < 8; ++r) {
        int m = m0 + ms * 16 + r + half * 8;
        float val = acc[ms][cs][r] + ((bias_axis == 0) ? bc : bias[m]);
        D[(size_t)m * Nout + col] = (bf16_t)val;
      }
    }
}

// ---- attention helpers ----
__device__ __forceinline__ void load_ktile(const bf16_t* const* krow, int nt, v16bf kb[2][2]) {
#pragma unroll
  for (int j = 0; j < 2; ++j) {
    const bf16_t* kr = krow[j] + (size_t)nt * DQV;
    kb[j][0] = load_op(kr,      kr + 8);
    kb[j][1] = load_op(kr + 32, kr + 40);
  }
}

__device__ __forceinline__ void load_vtile(const bf16_t* const* vrow, int nt, v16bf vb[4]) {
#pragma unroll
  for (int cs = 0; cs < 4; ++cs) vb[cs] = load_op(vrow[cs] + nt, vrow[cs] + nt + 8);
}

// One 32-key flash-attention step: S = QK^T, online softmax, O += P V.
__device__ __forceinline__ void attn_stage(const v16bf kb[2][2], const v16bf vb[4],
                                           v16bf qa0, v16bf qa1,
                                           v8f* acc, float* rmax, float* rsum,
                                           bf16_t* ldsPw, int half, int lm) {
  v8f S[2];
#pragma unroll
  for (int j = 0; j < 2; ++j) {
    v8f s = wmma_bf16(qa0, kb[j][0], zero8());
    S[j]  = wmma_bf16(qa1, kb[j][1], s);
  }
  float scl[8];
#pragma unroll
  for (int r = 0; r < 8; ++r) {
    float t = fmaxf(S[0][r], S[1][r]);
    t = fmaxf(t, __shfl_xor(t, 1));
    t = fmaxf(t, __shfl_xor(t, 2));
    t = fmaxf(t, __shfl_xor(t, 4));
    t = fmaxf(t, __shfl_xor(t, 8));
    float nm = fmaxf(rmax[r], t);
    scl[r] = __expf(rmax[r] - nm);
    rmax[r] = nm;
    float p0 = __expf(S[0][r] - nm);
    float p1 = __expf(S[1][r] - nm);
    S[0][r] = p0; S[1][r] = p1;
    float ts = p0 + p1;
    ts += __shfl_xor(ts, 1);
    ts += __shfl_xor(ts, 2);
    ts += __shfl_xor(ts, 4);
    ts += __shfl_xor(ts, 8);
    rsum[r] = rsum[r] * scl[r] + ts;
  }
#pragma unroll
  for (int cs = 0; cs < 4; ++cs)
#pragma unroll
    for (int r = 0; r < 8; ++r) acc[cs][r] *= scl[r];

  // transpose P through wave-private LDS: D-layout -> A-layout
#pragma unroll
  for (int r = 0; r < 8; ++r) {
    ldsPw[(r + 8 * half) * 32 + lm]      = (bf16_t)S[0][r];
    ldsPw[(r + 8 * half) * 32 + 16 + lm] = (bf16_t)S[1][r];
  }
  v16bf pa = load_op(&ldsPw[lm * 32 + half * 16],
                     &ldsPw[lm * 32 + 8 + half * 16]);
#pragma unroll
  for (int cs = 0; cs < 4; ++cs) acc[cs] = wmma_bf16(pa, vb[cs], acc[cs]);
}

// ---------------- streaming (flash) cross-attention ----------------
// Q,K token-major [B][N][DQ] bf16; V channel-major [B][C][N] bf16.
// out[c][m] = x[c][m] + gamma * sum_n softmax_n(Q[m].K[n]) * V[c][n]
// Two-stage ping-pong over 32-key tiles: each tile's loads are in flight across
// a full WMMA+softmax stage before first use.
__global__ __launch_bounds__(256) void k_attn(
    const bf16_t* __restrict__ Q1, const bf16_t* __restrict__ K1, const bf16_t* __restrict__ V1,
    const bf16_t* __restrict__ Q2, const bf16_t* __restrict__ K2, const bf16_t* __restrict__ V2,
    const float* __restrict__ x1, const float* __restrict__ x2,
    const float* __restrict__ g1, const float* __restrict__ g2,
    float* __restrict__ out) {
  __shared__ bf16_t ldsP[8][16 * 32];  // per-wave P transpose buffer

  const int b    = blockIdx.z;
  const int dir  = blockIdx.y;
  const int m0   = blockIdx.x * 16;
  const int wave = threadIdx.x >> 5;
  const int lane = threadIdx.x & 31;
  const int half = lane >> 4, lm = lane & 15;
  const int c0   = wave * 64;  // 8 waves * 64 channels = 512

  const bf16_t* Q = dir ? Q2 : Q1;
  const bf16_t* K = dir ? K1 : K2;
  const bf16_t* V = dir ? V1 : V2;
  const float*  X = dir ? x2 : x1;
  const float gamma = dir ? g2[0] : g1[0];
  float* O = out + (size_t)dir * BB * CC * NN;

  // Preload Q A-operands (loop invariant)
  const bf16_t* qrow = Q + ((size_t)b * NN + m0 + lm) * DQV;
  v16bf qa0 = load_op(qrow + half * 16,      qrow + 8 + half * 16);  // d 0..31
  v16bf qa1 = load_op(qrow + 32 + half * 16, qrow + 40 + half * 16);  // d 32..63

  const bf16_t* Kbase = K + (size_t)b * NN * DQV;
  const bf16_t* Vbase = V + (size_t)b * CC * NN;
  const bf16_t* krow[2];
#pragma unroll
  for (int j = 0; j < 2; ++j)
    krow[j] = Kbase + (size_t)(j * 16 + lm) * DQV + half * 16;
  const bf16_t* vrow[4];
#pragma unroll
  for (int cs = 0; cs < 4; ++cs)
    vrow[cs] = Vbase + (size_t)(c0 + cs * 16 + lm) * NN + half * 16;

  v8f acc[4];
#pragma unroll
  for (int cs = 0; cs < 4; ++cs) acc[cs] = zero8();
  float rmax[8], rsum[8];
#pragma unroll
  for (int r = 0; r < 8; ++r) { rmax[r] = -1.0e30f; rsum[r] = 0.0f; }

  bf16_t* ldsPw = &ldsP[wave][0];

  v16bf kbA[2][2], vbA[4], kbB[2][2], vbB[4];
  load_ktile(krow, 0, kbA);
  load_vtile(vrow, 0, vbA);

  for (int nt = 0; nt < NN; nt += 64) {
    // stage 0: consume tile nt (A buffers); issue tile nt+32 (B buffers)
    load_ktile(krow, nt + 32, kbB);
    load_vtile(vrow, nt + 32, vbB);
    if (nt + 256 < NN) {  // long-range prefetch to pre-warm the K/V streams
      __builtin_prefetch(krow[0] + (size_t)(nt + 256) * DQV, 0, 1);
#pragma unroll
      for (int cs = 0; cs < 4; ++cs)
        __builtin_prefetch(vrow[cs] + nt + 256, 0, 1);
    }
    attn_stage(kbA, vbA, qa0, qa1, acc, rmax, rsum, ldsPw, half, lm);

    // stage 1: consume tile nt+32 (B buffers); issue tile nt+64 (A buffers)
    int nt2 = (nt + 64 < NN) ? nt + 64 : 0;  // clamp: final prefetch discarded
    load_ktile(krow, nt2, kbA);
    load_vtile(vrow, nt2, vbA);
    attn_stage(kbB, vbB, qa0, qa1, acc, rmax, rsum, ldsPw, half, lm);
  }

  // ---- epilogue: out = x + gamma * O / rowsum ----
  float giv[8];
#pragma unroll
  for (int r = 0; r < 8; ++r) giv[r] = gamma / rsum[r];
#pragma unroll
  for (int cs = 0; cs < 4; ++cs) {
    int c = c0 + cs * 16 + lm;
    size_t base = ((size_t)b * CC + c) * NN + m0 + half * 8;  // rows r = consecutive tokens
    f32x4 xv0 = *(const f32x4*)(X + base);
    f32x4 xv1 = *(const f32x4*)(X + base + 4);
    f32x4 o0, o1;
#pragma unroll
    for (int r = 0; r < 4; ++r) o0[r] = xv0[r] + acc[cs][r] * giv[r];
#pragma unroll
    for (int r = 0; r < 4; ++r) o1[r] = xv1[r] + acc[cs][4 + r] * giv[4 + r];
    *(f32x4*)(O + base)     = o0;
    *(f32x4*)(O + base + 4) = o1;
  }
}

extern "C" void kernel_launch(void* const* d_in, const int* in_sizes, int n_in,
                              void* d_out, int out_size, void* d_ws, size_t ws_size,
                              hipStream_t stream) {
  (void)in_sizes; (void)n_in; (void)out_size; (void)ws_size;
  const float* x1  = (const float*)d_in[0];
  const float* x2  = (const float*)d_in[1];
  const float* wq1 = (const float*)d_in[2];
  const float* bq1 = (const float*)d_in[3];
  const float* wk1 = (const float*)d_in[4];
  const float* bk1 = (const float*)d_in[5];
  const float* wv1 = (const float*)d_in[6];
  const float* bv1 = (const float*)d_in[7];
  const float* wq2 = (const float*)d_in[8];
  const float* bq2 = (const float*)d_in[9];
  const float* wk2 = (const float*)d_in[10];
  const float* bk2 = (const float*)d_in[11];
  const float* wv2 = (const float*)d_in[12];
  const float* bv2 = (const float*)d_in[13];
  const float* g1  = (const float*)d_in[14];
  const float* g2  = (const float*)d_in[15];

  bf16_t* ws = (bf16_t*)d_ws;
  const size_t SZ_XT = (size_t)BB * NN * CC;
  const size_t SZ_QK = (size_t)BB * NN * DQV;
  const size_t SZ_V  = (size_t)BB * CC * NN;
  bf16_t* Xt1  = ws;
  bf16_t* Xt2  = Xt1 + SZ_XT;
  bf16_t* Q1   = Xt2 + SZ_XT;
  bf16_t* K1   = Q1 + SZ_QK;
  bf16_t* Q2   = K1 + SZ_QK;
  bf16_t* K2   = Q2 + SZ_QK;
  bf16_t* V1   = K2 + SZ_QK;
  bf16_t* V2   = V1 + SZ_V;
  bf16_t* wq1b = V2 + SZ_V;
  bf16_t* wk1b = wq1b + (size_t)DQV * CC;
  bf16_t* wq2b = wk1b + (size_t)DQV * CC;
  bf16_t* wk2b = wq2b + (size_t)DQV * CC;
  bf16_t* wv1b = wk2b + (size_t)DQV * CC;
  bf16_t* wv2b = wv1b + (size_t)CC * CC;

  // 1) weights -> bf16
  k_cvt_bf16<<<(DQV * CC + 255) / 256, 256, 0, stream>>>(wq1, wq1b, DQV * CC);
  k_cvt_bf16<<<(DQV * CC + 255) / 256, 256, 0, stream>>>(wk1, wk1b, DQV * CC);
  k_cvt_bf16<<<(DQV * CC + 255) / 256, 256, 0, stream>>>(wq2, wq2b, DQV * CC);
  k_cvt_bf16<<<(DQV * CC + 255) / 256, 256, 0, stream>>>(wk2, wk2b, DQV * CC);
  k_cvt_bf16<<<(CC * CC + 255) / 256, 256, 0, stream>>>(wv1, wv1b, CC * CC);
  k_cvt_bf16<<<(CC * CC + 255) / 256, 256, 0, stream>>>(wv2, wv2b, CC * CC);

  // 2) x -> token-major bf16
  dim3 tg(NN / 32, CC / 32, BB);
  k_transpose_cvt<<<tg, 256, 0, stream>>>(x1, Xt1);
  k_transpose_cvt<<<tg, 256, 0, stream>>>(x2, Xt2);

  // 3) Q/K projections: D[token][d] = Xt * W^T  (token-major output)
  dim3 qg((NN / 32) * (DQV / 64) / 4, BB, 1);
  k_gemm_tn<<<qg, 128, 0, stream>>>(Xt1, (size_t)NN * CC, wq1b, 0, bq1,
                                    Q1, (size_t)NN * DQV, NN, DQV, CC, 0);
  k_gemm_tn<<<qg, 128, 0, stream>>>(Xt1, (size_t)NN * CC, wk1b, 0, bk1,
                                    K1, (size_t)NN * DQV, NN, DQV, CC, 0);
  k_gemm_tn<<<qg, 128, 0, stream>>>(Xt2, (size_t)NN * CC, wq2b, 0, bq2,
                                    Q2, (size_t)NN * DQV, NN, DQV, CC, 0);
  k_gemm_tn<<<qg, 128, 0, stream>>>(Xt2, (size_t)NN * CC, wk2b, 0, bk2,
                                    K2, (size_t)NN * DQV, NN, DQV, CC, 0);

  // 4) V projections flipped: D[c_out][token] = Wv * X  (channel-major output)
  dim3 vg((CC / 32) * (NN / 64) / 4, BB, 1);
  k_gemm_tn<<<vg, 128, 0, stream>>>(wv1b, 0, Xt1, (size_t)NN * CC, bv1,
                                    V1, (size_t)CC * NN, CC, NN, CC, 1);
  k_gemm_tn<<<vg, 128, 0, stream>>>(wv2b, 0, Xt2, (size_t)NN * CC, bv2,
                                    V2, (size_t)CC * NN, CC, NN, CC, 1);

  // 5) streaming cross-attention, both directions
  dim3 ag(NN / 16, 2, BB);
  k_attn<<<ag, 256, 0, stream>>>(Q1, K1, V1, Q2, K2, V2, x1, x2, g1, g2,
                                 (float*)d_out);
}
